// ODEModel_71932112273992
// MI455X (gfx1250) — matrix operand order repaired and checked by reading, not compiled, not run on hardware
//
#include <hip/hip_runtime.h>
#include <hip/hip_bf16.h>
#include <math.h>

// Problem constants (match reference)
#define BB     4096
#define OBS    128
#define LAT    512
#define DH     2048
#define OUTD   64
#define NSTEPS 20

typedef __attribute__((ext_vector_type(16))) __bf16 bf16x16;
typedef __attribute__((ext_vector_type(8)))  float  f32x8;

// Hardware tanh on gfx1250 (V_TANH_F32, single TRANS-pipe instruction that
// co-executes with the WMMA/XDL pipe). Fall back to libm if the builtin is
// not declared by this toolchain.
#if __has_builtin(__builtin_amdgcn_tanhf)
__device__ __forceinline__ float fast_tanh(float x) {
    return __builtin_amdgcn_tanhf(x);
}
#else
__device__ __forceinline__ float fast_tanh(float x) { return tanhf(x); }
#endif

// ---------------------------------------------------------------------------
// Fragment loaders (CDNA5 WMMA 16x16x32 bf16 layouts, ISA 7.12.2)
// A (16-bit, 16x32): lane<16 -> row M=lane, K = {0..7, 16..23};
//                    lane>=16 -> row M=lane-16, K = {8..15, 24..31}
// B from pre-transposed weights BT[N][K]: lane<16 -> col N=lane, K = {0..7, 16..23};
//                    lane>=16 -> col N=lane-16, K = {8..15, 24..31} (matches A)
// C/D (f32): lane<16 -> col N=lane rows M=0..7 ; lane>=16 -> col N=lane-16 rows 8..15
// ---------------------------------------------------------------------------
__device__ __forceinline__ bf16x16 load_a_frag(const __bf16* A, int row, int K,
                                               int k0, int lane) {
    union { bf16x16 v; float4 q[2]; } u;
    const int kh = (lane >> 4) * 8;
    const __bf16* p = A + (size_t)row * K + k0 + kh;
    u.q[0] = *(const float4*)(p);
    u.q[1] = *(const float4*)(p + 16);
    return u.v;
}

__device__ __forceinline__ bf16x16 load_b_frag(const __bf16* BT, int col, int K,
                                               int k0, int lane) {
    union { bf16x16 v; float4 q[2]; } u;
    const int ks = k0 + (lane >> 4) * 8;
    const __bf16* p = BT + (size_t)col * K + ks;
    u.q[0] = *(const float4*)(p);
    u.q[1] = *(const float4*)(p + 16);
    return u.v;
}

// 64x64 tile per wave, K-loop stepping 32, 16 v_wmma per step.
template <int K>
__device__ __forceinline__ void wave_gemm64(const __bf16* A, const __bf16* BT,
                                            int m0, int n0, int lane,
                                            f32x8 C[4][4]) {
#pragma unroll 1
    for (int k0 = 0; k0 < K; k0 += 32) {
        bf16x16 a[4], b[4];
#pragma unroll
        for (int i = 0; i < 4; ++i)
            a[i] = load_a_frag(A, m0 + i * 16 + (lane & 15), K, k0, lane);
#pragma unroll
        for (int i = 0; i < 4; ++i)
            b[i] = load_b_frag(BT, n0 + i * 16 + (lane & 15), K, k0, lane);
#pragma unroll
        for (int mi = 0; mi < 4; ++mi)
#pragma unroll
            for (int ni = 0; ni < 4; ++ni)
                C[mi][ni] = __builtin_amdgcn_wmma_f32_16x16x32_bf16(
                    false, a[mi], false, b[ni], (short)0, C[mi][ni], false, false);
    }
}

// ---------------------------------------------------------------------------
// h0 = tanh(x @ W_in + b_in)   -> h (f32) and hin (bf16)
// ---------------------------------------------------------------------------
__global__ __launch_bounds__(256) void k_h0(const __bf16* __restrict__ xb,
                                            const __bf16* __restrict__ WinT,
                                            const float* __restrict__ b_in,
                                            float* __restrict__ h,
                                            __bf16* __restrict__ hin) {
    const int lane = threadIdx.x & 31;
    const int w = blockIdx.x * 8 + (threadIdx.x >> 5);
    const int TN = LAT / 64;
    const int m0 = (w / TN) * 64, n0 = (w % TN) * 64;
    f32x8 C[4][4] = {};
    wave_gemm64<OBS>(xb, WinT, m0, n0, lane, C);
#pragma unroll
    for (int ni = 0; ni < 4; ++ni) {
        const int col = n0 + ni * 16 + (lane & 15);
        const float bias = b_in[col];
#pragma unroll
        for (int mi = 0; mi < 4; ++mi) {
            const int rbase = m0 + mi * 16 + (lane >> 4) * 8;
#pragma unroll
            for (int r = 0; r < 8; ++r) {
                const float v = fast_tanh(C[mi][ni][r] + bias);
                const size_t idx = (size_t)(rbase + r) * LAT + col;
                h[idx] = v;
                hin[idx] = (__bf16)v;
            }
        }
    }
}

// ---------------------------------------------------------------------------
// GEMM1 of dynamics: z = tanh(hin @ W1[0:512] + (b1 + t*W1[512,:]))
// (the time-concat column is folded into the bias)
// ---------------------------------------------------------------------------
__global__ __launch_bounds__(256) void k_dyn1(const __bf16* __restrict__ hin,
                                              const __bf16* __restrict__ W1T,
                                              const float* __restrict__ b1,
                                              const float* __restrict__ trow,
                                              float t,
                                              __bf16* __restrict__ z) {
    const int lane = threadIdx.x & 31;
    const int w = blockIdx.x * 8 + (threadIdx.x >> 5);
    const int TN = DH / 64;
    const int m0 = (w / TN) * 64, n0 = (w % TN) * 64;
    f32x8 C[4][4] = {};
    wave_gemm64<LAT>(hin, W1T, m0, n0, lane, C);
#pragma unroll
    for (int ni = 0; ni < 4; ++ni) {
        const int col = n0 + ni * 16 + (lane & 15);
        const float bias = b1[col] + t * trow[col];
#pragma unroll
        for (int mi = 0; mi < 4; ++mi) {
            const int rbase = m0 + mi * 16 + (lane >> 4) * 8;
#pragma unroll
            for (int r = 0; r < 8; ++r) {
                const float v = fast_tanh(C[mi][ni][r] + bias);
                z[(size_t)(rbase + r) * DH + col] = (__bf16)v;
            }
        }
    }
}

// ---------------------------------------------------------------------------
// GEMM2 of dynamics + fused RK4 stage bookkeeping:
//   k = z @ W2 + b2
//   non-final stage s: acc = aPrev*acc + aK*k ; hin = bf16(h + hinK*k)
//   final stage:       h  += (dt/6)*(acc + k) ; hin = bf16(h)
// ---------------------------------------------------------------------------
__global__ __launch_bounds__(256) void k_dyn2(const __bf16* __restrict__ z,
                                              const __bf16* __restrict__ W2T,
                                              const float* __restrict__ b2,
                                              float* __restrict__ h,
                                              float* __restrict__ acc,
                                              __bf16* __restrict__ hin,
                                              float aPrev, float aK, float hinK,
                                              float finalScale, int isFinal) {
    const int lane = threadIdx.x & 31;
    const int w = blockIdx.x * 8 + (threadIdx.x >> 5);
    const int TN = LAT / 64;
    const int m0 = (w / TN) * 64, n0 = (w % TN) * 64;
    f32x8 C[4][4] = {};
    wave_gemm64<DH>(z, W2T, m0, n0, lane, C);
#pragma unroll
    for (int ni = 0; ni < 4; ++ni) {
        const int col = n0 + ni * 16 + (lane & 15);
        const float bias = b2[col];
#pragma unroll
        for (int mi = 0; mi < 4; ++mi) {
            const int rbase = m0 + mi * 16 + (lane >> 4) * 8;
#pragma unroll
            for (int r = 0; r < 8; ++r) {
                const float k = C[mi][ni][r] + bias;
                const size_t idx = (size_t)(rbase + r) * LAT + col;
                if (isFinal) {
                    const float hn = h[idx] + finalScale * (acc[idx] + k);
                    h[idx] = hn;
                    hin[idx] = (__bf16)hn;
                } else {
                    acc[idx] = aPrev * acc[idx] + aK * k;
                    hin[idx] = (__bf16)(h[idx] + hinK * k);
                }
            }
        }
    }
}

// ---------------------------------------------------------------------------
// out = hT @ W_out + b_out   (N = 64 -> one 64-wide tile column)
// ---------------------------------------------------------------------------
__global__ __launch_bounds__(256) void k_out(const __bf16* __restrict__ hin,
                                             const __bf16* __restrict__ WoutT,
                                             const float* __restrict__ b_out,
                                             float* __restrict__ out) {
    const int lane = threadIdx.x & 31;
    const int w = blockIdx.x * 8 + (threadIdx.x >> 5);
    const int m0 = w * 64, n0 = 0;
    f32x8 C[4][4] = {};
    wave_gemm64<LAT>(hin, WoutT, m0, n0, lane, C);
#pragma unroll
    for (int ni = 0; ni < 4; ++ni) {
        const int col = n0 + ni * 16 + (lane & 15);
        const float bias = b_out[col];
#pragma unroll
        for (int mi = 0; mi < 4; ++mi) {
            const int rbase = m0 + mi * 16 + (lane >> 4) * 8;
#pragma unroll
            for (int r = 0; r < 8; ++r)
                out[(size_t)(rbase + r) * OUTD + col] = C[mi][ni][r] + bias;
        }
    }
}

// ---------------------------------------------------------------------------
// Conversion helpers
// ---------------------------------------------------------------------------
__global__ void k_cvt(const float* __restrict__ s, __bf16* __restrict__ d, int n) {
    int i = blockIdx.x * blockDim.x + threadIdx.x;
    if (i < n) d[i] = (__bf16)s[i];
}

// dst[n*K + k] = src[k*N + n]   (src row-major [K,N] f32 -> dst [N,K] bf16)
__global__ void k_cvt_tr(const float* __restrict__ src, __bf16* __restrict__ dst,
                         int K, int N) {
    int i = blockIdx.x * blockDim.x + threadIdx.x;
    if (i < K * N) {
        int n = i / K, k = i % K;
        dst[i] = (__bf16)src[(size_t)k * N + n];
    }
}

__global__ void k_trow(const float* __restrict__ W1, float* __restrict__ trow) {
    int n = blockIdx.x * blockDim.x + threadIdx.x;
    if (n < DH) trow[n] = W1[(size_t)LAT * DH + n];   // row 512 of W1 (the t column)
}

// ---------------------------------------------------------------------------
extern "C" void kernel_launch(void* const* d_in, const int* in_sizes, int n_in,
                              void* d_out, int out_size, void* d_ws, size_t ws_size,
                              hipStream_t stream) {
    const float* x     = (const float*)d_in[0];
    const float* W_in  = (const float*)d_in[1];
    const float* b_in  = (const float*)d_in[2];
    const float* W1    = (const float*)d_in[3];
    const float* b1    = (const float*)d_in[4];
    const float* W2    = (const float*)d_in[5];
    const float* b2    = (const float*)d_in[6];
    const float* W_out = (const float*)d_in[7];
    const float* b_out = (const float*)d_in[8];
    float* out = (float*)d_out;

    char* ws = (char*)d_ws;
    size_t off = 0;
    auto alloc = [&](size_t bytes) -> char* {
        char* p = ws + off;
        off += (bytes + 255) & ~(size_t)255;
        return p;
    };
    float*  h     = (float*) alloc((size_t)BB * LAT * 4);
    float*  acc   = (float*) alloc((size_t)BB * LAT * 4);
    __bf16* hin   = (__bf16*)alloc((size_t)BB * LAT * 2);
    __bf16* z     = (__bf16*)alloc((size_t)BB * DH * 2);
    __bf16* xb    = (__bf16*)alloc((size_t)BB * OBS * 2);
    __bf16* W1T   = (__bf16*)alloc((size_t)DH * LAT * 2);
    float*  trow  = (float*) alloc((size_t)DH * 4);
    __bf16* W2T   = (__bf16*)alloc((size_t)LAT * DH * 2);
    __bf16* WinT  = (__bf16*)alloc((size_t)LAT * OBS * 2);
    __bf16* WoutT = (__bf16*)alloc((size_t)OUTD * LAT * 2);

    // One-time conversions (weights + input to bf16; transposed to [N,K])
    k_cvt   <<<(BB * OBS + 255) / 256, 256, 0, stream>>>(x, xb, BB * OBS);
    k_cvt_tr<<<(LAT * DH + 255) / 256, 256, 0, stream>>>(W1, W1T, LAT, DH);
    k_trow  <<<(DH + 255) / 256, 256, 0, stream>>>(W1, trow);
    k_cvt_tr<<<(DH * LAT + 255) / 256, 256, 0, stream>>>(W2, W2T, DH, LAT);
    k_cvt_tr<<<(OBS * LAT + 255) / 256, 256, 0, stream>>>(W_in, WinT, OBS, LAT);
    k_cvt_tr<<<(LAT * OUTD + 255) / 256, 256, 0, stream>>>(W_out, WoutT, LAT, OUTD);

    // h0 = tanh(x @ W_in + b_in)
    k_h0<<<(BB / 64) * (LAT / 64) / 8, 256, 0, stream>>>(xb, WinT, b_in, h, hin);

    const float dt = 1.0f / (float)NSTEPS;
    const int g1 = (BB / 64) * (DH / 64) / 8;   // 256 blocks
    const int g2 = (BB / 64) * (LAT / 64) / 8;  // 64 blocks

    for (int s = 0; s < NSTEPS; ++s) {
        const float t = dt * (float)s;
        // k1
        k_dyn1<<<g1, 256, 0, stream>>>(hin, W1T, b1, trow, t, z);
        k_dyn2<<<g2, 256, 0, stream>>>(z, W2T, b2, h, acc, hin,
                                       0.0f, 1.0f, 0.5f * dt, 0.0f, 0);
        // k2
        k_dyn1<<<g1, 256, 0, stream>>>(hin, W1T, b1, trow, t + 0.5f * dt, z);
        k_dyn2<<<g2, 256, 0, stream>>>(z, W2T, b2, h, acc, hin,
                                       1.0f, 2.0f, 0.5f * dt, 0.0f, 0);
        // k3
        k_dyn1<<<g1, 256, 0, stream>>>(hin, W1T, b1, trow, t + 0.5f * dt, z);
        k_dyn2<<<g2, 256, 0, stream>>>(z, W2T, b2, h, acc, hin,
                                       1.0f, 2.0f, dt, 0.0f, 0);
        // k4 + h update
        k_dyn1<<<g1, 256, 0, stream>>>(hin, W1T, b1, trow, t + dt, z);
        k_dyn2<<<g2, 256, 0, stream>>>(z, W2T, b2, h, acc, hin,
                                       0.0f, 0.0f, 0.0f, dt / 6.0f, 1);
    }

    // out = hT @ W_out + b_out
    k_out<<<(BB / 64) / 8, 256, 0, stream>>>(hin, WoutT, b_out, out);
}
